// DendriticMLP_6657199309465
// MI455X (gfx1250) — compile-verified, matrix operand order checked by reading, not run on hardware
//
#include <hip/hip_runtime.h>
#include <math.h>

// ---------------------------------------------------------------------------
// CDNA5 (gfx1250) wave32 WMMA types
// ---------------------------------------------------------------------------
typedef __attribute__((ext_vector_type(16))) __bf16       v16bf;
typedef __attribute__((ext_vector_type(8)))  float        v8f;
typedef __attribute__((ext_vector_type(8)))  unsigned int v8u;
typedef __attribute__((ext_vector_type(4)))  unsigned int v4u;

// ---- scalar / packed f32 <-> bf16 helpers (used OUTSIDE hot loops only) ----
__device__ __forceinline__ unsigned int pk_bf16(float lo, float hi) {
  unsigned int ul = __builtin_bit_cast(unsigned int, lo);
  unsigned int uh = __builtin_bit_cast(unsigned int, hi);
  ul = (ul + 0x7FFFu + ((ul >> 16) & 1u)) >> 16;   // RNE
  uh = (uh + 0x7FFFu + ((uh >> 16) & 1u)) >> 16;
  return ul | (uh << 16);
}
__device__ __forceinline__ float bf16_to_f32(unsigned short s) {
  unsigned int u = ((unsigned int)s) << 16;
  return __builtin_bit_cast(float, u);
}

// Load one 16x32 (A) / 32x16 (B) bf16 WMMA fragment. `base` is the per-lane
// pointer (row + k-half already applied): elements [k..k+7] and [k+16..k+23].
__device__ __forceinline__ v16bf load_frag(const __bf16* __restrict__ base,
                                           int k) {
  const v4u* q = (const v4u*)(base + k);
  v4u lo = q[0];   // k+0..7
  v4u hi = q[2];   // k+16..23
  v8u u;
  u[0] = lo[0]; u[1] = lo[1]; u[2] = lo[2]; u[3] = lo[3];
  u[4] = hi[0]; u[5] = hi[1]; u[6] = hi[2]; u[7] = hi[3];
  return __builtin_bit_cast(v16bf, u);
}

__device__ __forceinline__ v8f wmma_bf16(v16bf a, v16bf b, v8f c) {
  return __builtin_amdgcn_wmma_f32_16x16x32_bf16(
      /*neg_a=*/false, a, /*neg_b=*/false, b,
      /*c_mod=*/(short)0, c, /*reuse_a=*/false, /*reuse_b=*/false);
}

// One pipeline stage: A0/A1 row fragments + 4 B fragments (wave tile 32x64).
struct FragSet {
  v16bf a0, a1, b0, b1, b2, b3;
};

__device__ __forceinline__ void load_set(FragSet& f,
                                         const __bf16* __restrict__ pa0,
                                         const __bf16* __restrict__ pa1,
                                         const __bf16* __restrict__ pw0,
                                         const __bf16* __restrict__ pw1,
                                         const __bf16* __restrict__ pw2,
                                         const __bf16* __restrict__ pw3,
                                         int k) {
  f.a0 = load_frag(pa0, k);
  f.a1 = load_frag(pa1, k);
  f.b0 = load_frag(pw0, k);
  f.b1 = load_frag(pw1, k);
  f.b2 = load_frag(pw2, k);
  f.b3 = load_frag(pw3, k);
}

__device__ __forceinline__ void mma_set(v8f (&acc)[2][4], const FragSet& f) {
  acc[0][0] = wmma_bf16(f.a0, f.b0, acc[0][0]);
  acc[1][0] = wmma_bf16(f.a1, f.b0, acc[1][0]);
  acc[0][1] = wmma_bf16(f.a0, f.b1, acc[0][1]);
  acc[1][1] = wmma_bf16(f.a1, f.b1, acc[1][1]);
  acc[0][2] = wmma_bf16(f.a0, f.b2, acc[0][2]);
  acc[1][2] = wmma_bf16(f.a1, f.b2, acc[1][2]);
  acc[0][3] = wmma_bf16(f.a0, f.b3, acc[0][3]);
  acc[1][3] = wmma_bf16(f.a1, f.b3, acc[1][3]);
}

// ---------------------------------------------------------------------------
// Elementwise f32 -> bf16 (RNE), 4 elements per thread. n4 = count/4.
// ---------------------------------------------------------------------------
__global__ __launch_bounds__(256) void dmlp_cvt_bf16(
    const float* __restrict__ src, __bf16* __restrict__ dst, int n4) {
  const int i = blockIdx.x * 256 + threadIdx.x;
  if (i < n4) {
    float4 f = ((const float4*)src)[i];
    uint2 o;
    o.x = pk_bf16(f.x, f.y);
    o.y = pk_bf16(f.z, f.w);
    ((uint2*)dst)[i] = o;
  }
}

// ---------------------------------------------------------------------------
// GEMM:  C(MxN) f32 = A(MxK) * W(NxK)^T + bias   (A, W bf16 in memory)
// Block = 4 waves (128 thr). Wave tile 32x64, block tile 128x64.
// Ping-pong double-buffered K loop (no register rotation copies).
// Requires K % 64 == 0 (true here: 1024 / 2048).
// ---------------------------------------------------------------------------
__global__ __launch_bounds__(128) void dmlp_gemm_bias(
    const __bf16* __restrict__ A, int lda,
    const __bf16* __restrict__ W,           // N x K, ld = K
    const float* __restrict__ bias,
    float* __restrict__ C, int ldcc,
    int N, int K) {
  const int wave = threadIdx.x >> 5;
  const int lane = threadIdx.x & 31;
  const int hh   = lane >> 4;
  const int ln   = lane & 15;
  const int m0   = blockIdx.x * 128 + wave * 32;
  const int n0   = blockIdx.y * 64;

  const __bf16* pa0 = A + (size_t)(m0 + ln) * lda + hh * 8;
  const __bf16* pa1 = pa0 + (size_t)16 * lda;
  int nr[4];
#pragma unroll
  for (int j = 0; j < 4; ++j) {
    nr[j] = n0 + j * 16 + ln;
    if (nr[j] >= N) nr[j] = N - 1;          // clamp (store is guarded)
  }
  const __bf16* pw0 = W + (size_t)nr[0] * K + hh * 8;
  const __bf16* pw1 = W + (size_t)nr[1] * K + hh * 8;
  const __bf16* pw2 = W + (size_t)nr[2] * K + hh * 8;
  const __bf16* pw3 = W + (size_t)nr[3] * K + hh * 8;

  v8f acc[2][4] = {};
  FragSet s0, s1;

  load_set(s0, pa0, pa1, pw0, pw1, pw2, pw3, 0);
  int k = 32;
  for (; k + 32 < K; k += 64) {
    load_set(s1, pa0, pa1, pw0, pw1, pw2, pw3, k);
    mma_set(acc, s0);
    load_set(s0, pa0, pa1, pw0, pw1, pw2, pw3, k + 32);
    mma_set(acc, s1);
  }
  load_set(s1, pa0, pa1, pw0, pw1, pw2, pw3, k);   // k == K-32
  mma_set(acc, s0);
  mma_set(acc, s1);

#pragma unroll
  for (int i = 0; i < 2; ++i) {
#pragma unroll
    for (int j = 0; j < 4; ++j) {
      const int n = n0 + j * 16 + ln;
      if (n < N) {
        const float bv = bias[n];
#pragma unroll
        for (int r = 0; r < 8; ++r) {
          const int m = m0 + i * 16 + hh * 8 + r;
          C[(size_t)m * ldcc + n] = acc[i][j][r] + bv;
        }
      }
    }
  }
}

// ---------------------------------------------------------------------------
// Per-column mean / rstd over M rows (batchnorm stats, biased variance).
// ---------------------------------------------------------------------------
__global__ __launch_bounds__(256) void dmlp_stats(
    const float* __restrict__ Y, float* __restrict__ mean,
    float* __restrict__ rstd, int M, int N) {
  __shared__ float ssum[256];
  __shared__ float ssq[256];
  const int c   = threadIdx.x & 63;
  const int g   = threadIdx.x >> 6;
  const int col = blockIdx.x * 64 + c;
  float s = 0.f, s2 = 0.f;
  for (int m = g; m < M; m += 4) {
    float v = Y[(size_t)m * N + col];
    s += v;
    s2 += v * v;
  }
  ssum[threadIdx.x] = s;
  ssq[threadIdx.x]  = s2;
  __syncthreads();
  if (g == 0) {
    s  = ssum[c] + ssum[c + 64] + ssum[c + 128] + ssum[c + 192];
    s2 = ssq[c]  + ssq[c + 64]  + ssq[c + 128]  + ssq[c + 192];
    const float mu  = s / (float)M;
    const float var = s2 / (float)M - mu * mu;
    mean[col] = mu;
    rstd[col] = rsqrtf(var + 1e-5f);
  }
}

// ---------------------------------------------------------------------------
// h = relu((y - mean) * rstd), written as bf16. 4 elems/thread.
// ---------------------------------------------------------------------------
__global__ __launch_bounds__(256) void dmlp_bnrelu(
    const float* __restrict__ Y, const float* __restrict__ mean,
    const float* __restrict__ rstd, __bf16* __restrict__ H,
    int total4, int nmask) {
  const int i = blockIdx.x * 256 + threadIdx.x;
  if (i < total4) {
    float4 f = ((const float4*)Y)[i];
    const int c = (i * 4) & nmask;
    float v0 = (f.x - mean[c + 0]) * rstd[c + 0];
    float v1 = (f.y - mean[c + 1]) * rstd[c + 1];
    float v2 = (f.z - mean[c + 2]) * rstd[c + 2];
    float v3 = (f.w - mean[c + 3]) * rstd[c + 3];
    v0 = v0 > 0.f ? v0 : 0.f;
    v1 = v1 > 0.f ? v1 : 0.f;
    v2 = v2 > 0.f ? v2 : 0.f;
    v3 = v3 > 0.f ? v3 : 0.f;
    uint2 o;
    o.x = pk_bf16(v0, v1);
    o.y = pk_bf16(v2, v3);
    ((uint2*)H)[i] = o;
  }
}

// ---------------------------------------------------------------------------
// Dendrite stage, fused:
//   d[b, u*16+s] = ctx(b,:) . sw(u*16+s,:)       (WMMA, N = Hdim*16)
//   sel[b,u] = d value with max |.| over s       (shfl_xor in 16-lane half)
//   Hout[b,u] = Hin[b,u] * sigmoid(sel)          (bf16 in / bf16 out)
// Same ping-pong pipelined K-loop as the GEMM.
// ---------------------------------------------------------------------------
__global__ __launch_bounds__(128) void dmlp_dendrite(
    const __bf16* __restrict__ Ctx, int ldctx,
    const __bf16* __restrict__ SW,          // (Hdim*16) x K, ld = K
    const __bf16* __restrict__ Hin,
    __bf16* __restrict__ Hout,
    int Hdim, int K) {
  const int wave = threadIdx.x >> 5;
  const int lane = threadIdx.x & 31;
  const int hh   = lane >> 4;
  const int ln   = lane & 15;
  const int m0   = blockIdx.x * 128 + wave * 32;
  const int n0   = blockIdx.y * 64;          // 4 u-groups of 16 s

  const __bf16* pa0 = Ctx + (size_t)(m0 + ln) * ldctx + hh * 8;
  const __bf16* pa1 = pa0 + (size_t)16 * ldctx;
  const __bf16* pw0 = SW + (size_t)(n0 + 0 * 16 + ln) * K + hh * 8;
  const __bf16* pw1 = SW + (size_t)(n0 + 1 * 16 + ln) * K + hh * 8;
  const __bf16* pw2 = SW + (size_t)(n0 + 2 * 16 + ln) * K + hh * 8;
  const __bf16* pw3 = SW + (size_t)(n0 + 3 * 16 + ln) * K + hh * 8;

  v8f acc[2][4] = {};
  FragSet s0, s1;

  load_set(s0, pa0, pa1, pw0, pw1, pw2, pw3, 0);
  int k = 32;
  for (; k + 32 < K; k += 64) {
    load_set(s1, pa0, pa1, pw0, pw1, pw2, pw3, k);
    mma_set(acc, s0);
    load_set(s0, pa0, pa1, pw0, pw1, pw2, pw3, k + 32);
    mma_set(acc, s1);
  }
  load_set(s1, pa0, pa1, pw0, pw1, pw2, pw3, k);   // k == K-32
  mma_set(acc, s0);
  mma_set(acc, s1);

  const unsigned short* HinU = (const unsigned short*)Hin;
  unsigned short* HoutU = (unsigned short*)Hout;

#pragma unroll
  for (int i = 0; i < 2; ++i) {
#pragma unroll
    for (int j = 0; j < 4; ++j) {
      const int u = (n0 >> 4) + j;
#pragma unroll
      for (int r = 0; r < 8; ++r) {
        float v = acc[i][j][r];
        float a = fabsf(v);
#pragma unroll
        for (int mk = 1; mk < 16; mk <<= 1) {
          const float oa = __shfl_xor(a, mk, 32);
          const float ov = __shfl_xor(v, mk, 32);
          if (oa > a) { a = oa; v = ov; }
        }
        if (ln == 0) {
          const int m = m0 + i * 16 + hh * 8 + r;
          const float g = 1.f / (1.f + __expf(-v));
          const float hv = bf16_to_f32(HinU[(size_t)m * Hdim + u]) * g;
          HoutU[(size_t)m * Hdim + u] = (unsigned short)pk_bf16(hv, 0.f);
        }
      }
    }
  }
}

// ---------------------------------------------------------------------------
// Host-side launcher
// ---------------------------------------------------------------------------
extern "C" void kernel_launch(void* const* d_in, const int* in_sizes, int n_in,
                              void* d_out, int out_size, void* d_ws, size_t ws_size,
                              hipStream_t stream) {
  (void)in_sizes; (void)n_in; (void)out_size; (void)ws_size;

  constexpr int B   = 4096;
  constexpr int DIN = 1024;    // also D_CTX
  constexpr int H   = 2048;
  constexpr int OUT = 1000;
  constexpr size_t BH = (size_t)B * H;

  const float* x   = (const float*)d_in[0];    // B x 2048
  const float* w0  = (const float*)d_in[1];    // H x DIN
  const float* b0  = (const float*)d_in[2];
  const float* sw0 = (const float*)d_in[3];    // (H*16) x DIN
  const float* w1  = (const float*)d_in[4];    // H x H
  const float* b1  = (const float*)d_in[5];
  const float* sw1 = (const float*)d_in[6];
  const float* w2  = (const float*)d_in[7];
  const float* b2  = (const float*)d_in[8];
  const float* sw2 = (const float*)d_in[9];
  const float* wc  = (const float*)d_in[10];   // OUT x H
  const float* bc  = (const float*)d_in[11];
  float* out = (float*)d_out;                  // B x OUT

  // ---- workspace layout (~152 MB) ----
  char* w = (char*)d_ws;
  float*  Y    = (float*)w;            w += BH * sizeof(float);        // 32 MB
  __bf16* xbf  = (__bf16*)w;           w += BH * sizeof(__bf16);       // 16 MB
  __bf16* HR   = (__bf16*)w;           w += BH * sizeof(__bf16);       // 16 MB
  __bf16* Hd   = (__bf16*)w;           w += BH * sizeof(__bf16);       // 16 MB
  __bf16* wbf  = (__bf16*)w;           w += (size_t)H * H * sizeof(__bf16);        // 8 MB
  __bf16* swbf = (__bf16*)w;           w += (size_t)H * 16 * DIN * sizeof(__bf16); // 64 MB
  float*  mean = (float*)w;            w += H * sizeof(float);
  float*  rstd = (float*)w;

  const __bf16* ctx = xbf + DIN;               // B x DIN, ld 2048

  const dim3 blkG(128), blkS(256), blkE(256);
  const dim3 grdGemmH(B / 128, H / 64);                 // N = 2048
  const dim3 grdStats(H / 64);
  const dim3 grdBn((unsigned)(BH / 4 / 256));
  const dim3 grdDen(B / 128, (H * 16) / 64);            // N = 32768
  const dim3 grdCls(B / 128, (OUT + 63) / 64);          // N = 1000

  auto cvt = [&](const float* s, __bf16* d, size_t n) {
    const int n4 = (int)(n / 4);
    dmlp_cvt_bf16<<<dim3((n4 + 255) / 256), blkE, 0, stream>>>(s, d, n4);
  };

  // x (incl. ctx) -> bf16 once
  cvt(x, xbf, BH);

  // ---- layer 0 ----
  cvt(w0, wbf, (size_t)H * DIN);
  dmlp_gemm_bias<<<grdGemmH, blkG, 0, stream>>>(xbf, 2048, wbf, b0, Y, H, H, DIN);
  dmlp_stats<<<grdStats, blkS, 0, stream>>>(Y, mean, rstd, B, H);
  dmlp_bnrelu<<<grdBn, blkE, 0, stream>>>(Y, mean, rstd, HR, (int)(BH / 4), H - 1);
  cvt(sw0, swbf, (size_t)H * 16 * DIN);
  dmlp_dendrite<<<grdDen, blkG, 0, stream>>>(ctx, 2048, swbf, HR, Hd, H, DIN);

  // ---- layer 1 ----
  cvt(w1, wbf, (size_t)H * H);
  dmlp_gemm_bias<<<grdGemmH, blkG, 0, stream>>>(Hd, H, wbf, b1, Y, H, H, H);
  dmlp_stats<<<grdStats, blkS, 0, stream>>>(Y, mean, rstd, B, H);
  dmlp_bnrelu<<<grdBn, blkE, 0, stream>>>(Y, mean, rstd, HR, (int)(BH / 4), H - 1);
  cvt(sw1, swbf, (size_t)H * 16 * DIN);
  dmlp_dendrite<<<grdDen, blkG, 0, stream>>>(ctx, 2048, swbf, HR, Hd, H, DIN);

  // ---- layer 2 ----
  cvt(w2, wbf, (size_t)H * H);
  dmlp_gemm_bias<<<grdGemmH, blkG, 0, stream>>>(Hd, H, wbf, b2, Y, H, H, H);
  dmlp_stats<<<grdStats, blkS, 0, stream>>>(Y, mean, rstd, B, H);
  dmlp_bnrelu<<<grdBn, blkE, 0, stream>>>(Y, mean, rstd, HR, (int)(BH / 4), H - 1);
  cvt(sw2, swbf, (size_t)H * 16 * DIN);
  dmlp_dendrite<<<grdDen, blkG, 0, stream>>>(ctx, 2048, swbf, HR, Hd, H, DIN);

  // ---- classifier ----
  cvt(wc, wbf, (size_t)OUT * H);
  dmlp_gemm_bias<<<grdCls, blkG, 0, stream>>>(Hd, H, wbf, bc, out, OUT, OUT, H);
}
